// AdvancedGCN_45801531244897
// MI455X (gfx1250) — matrix operand order repaired and checked by reading, not compiled, not run on hardware
//
#include <hip/hip_runtime.h>
#include <hip/hip_bf16.h>
#include <math.h>
#include <string.h>

// ---------------- problem constants (fixed by the reference) ----------------
#define N_NODES 100000
#define E_EDGES 1600000
#define IN_F    128
#define H_F     64
#define MID_F   128
#define OUT_F   16
#define K1_KEEP 80000   // ceil(0.8 * 100000)
#define K2_KEEP 64000   // ceil(0.8 * 80000)

typedef __attribute__((ext_vector_type(16))) __bf16 v16bf;
typedef __attribute__((ext_vector_type(2)))  __bf16 v2bf;
typedef __attribute__((ext_vector_type(8)))  float  v8f;

struct U8 { unsigned x[8]; };   // 32 bytes == sizeof(v16bf)

// hardware bf16 conversion (clang lowers to v_cvt_*bf16* on gfx1250)
static __device__ __forceinline__ unsigned short f2bf(float f) {
  return __builtin_bit_cast(unsigned short, (__bf16)f);
}
static __device__ __forceinline__ unsigned pack2(float a, float b) {
  v2bf v; v[0] = (__bf16)a; v[1] = (__bf16)b;
  return __builtin_bit_cast(unsigned, v);
}
static __device__ __forceinline__ v16bf as_v16bf(const U8& u) {
  return __builtin_bit_cast(v16bf, u);
}
// order-preserving float <-> uint encoding (for atomic segment-max / radix select)
static __device__ __forceinline__ unsigned fenc(float f) {
  unsigned u = __float_as_uint(f);
  return (u & 0x80000000u) ? ~u : (u | 0x80000000u);
}
static __device__ __forceinline__ float fdec(unsigned u) {
  unsigned v = (u & 0x80000000u) ? (u & 0x7fffffffu) : ~u;
  return __uint_as_float(v);
}

// ---- Tensor Data Mover availability (device pass only; host pass -> fallback)
#if defined(__AMDGCN__) && __has_builtin(__builtin_amdgcn_tensor_load_to_lds) && \
    __has_builtin(__builtin_amdgcn_s_wait_tensorcnt)
#define HAVE_TDM 1
typedef __attribute__((ext_vector_type(4))) unsigned tdm_v4u;
typedef __attribute__((ext_vector_type(8))) int      tdm_v8i;
typedef __attribute__((ext_vector_type(4))) int      tdm_v4i;
typedef __attribute__((address_space(3))) unsigned short lds_us_t;
#else
#define HAVE_TDM 0
#endif

// ---------------- utility kernels ----------------
__global__ void k_fillu(unsigned* p, unsigned v, long n) {
  long t = (long)blockIdx.x * blockDim.x + threadIdx.x;
  if (t < n) p[t] = v;
}

// transpose + convert: W (K x Ncol f32, row-major) -> WT (Ncol x K bf16, row-major)
__global__ void k_transpose_bf16(const float* __restrict__ W, unsigned short* __restrict__ WT,
                                 int K, int Ncol) {
  int idx = blockIdx.x * blockDim.x + threadIdx.x;
  if (idx >= K * Ncol) return;
  int k = idx / Ncol, n = idx % Ncol;
  WT[n * K + k] = f2bf(W[(long)k * Ncol + n]);
}

// ---------------- node GEMM: Y(M x 64) = X(M x K) @ W(K x 64)  (no bias/relu) ----
// One wave handles 16 rows; 4 waves per block; WT (64 x K bf16) staged in LDS.
template <int K>
__global__ __launch_bounds__(128)
void k_gemm_node(const float* __restrict__ X, const unsigned short* __restrict__ WT,
                 float* __restrict__ Y, int M) {
  __shared__ unsigned short sWT[64 * K];
  for (int i = threadIdx.x; i < 64 * K; i += blockDim.x) sWT[i] = WT[i];
  __syncthreads();

  int wave = threadIdx.x >> 5, lane = threadIdx.x & 31;
  int r = lane & 15, hi = lane >> 4;
  int row16 = (blockIdx.x * 4 + wave) * 16;
  if (row16 >= M) return;
  int grow = row16 + r; if (grow >= M) grow = M - 1;
  const float* xr = X + (long)grow * K;

  constexpr int KC = K / 32;
  v16bf afrag[KC];
#pragma unroll
  for (int kc = 0; kc < KC; ++kc) {
    U8 u;
    int b0 = kc * 32 + 16 * hi, b1 = kc * 32 + 16 * hi + 8;
#pragma unroll
    for (int j = 0; j < 4; ++j) u.x[j]     = pack2(xr[b0 + 2 * j], xr[b0 + 2 * j + 1]);
#pragma unroll
    for (int j = 0; j < 4; ++j) u.x[4 + j] = pack2(xr[b1 + 2 * j], xr[b1 + 2 * j + 1]);
    afrag[kc] = as_v16bf(u);
  }
#pragma unroll
  for (int nt = 0; nt < 4; ++nt) {
    v8f c = {};
    U8 ub[2];
    {
      const unsigned* bp = (const unsigned*)&sWT[(nt * 16 + r) * K + 16 * hi];
#pragma unroll
      for (int j = 0; j < 8; ++j) ub[0].x[j] = bp[j];
    }
#pragma unroll
    for (int kc = 0; kc < KC; ++kc) {
      if (kc + 1 < KC) {   // double-buffer: issue next B-frag loads before the WMMA
        const unsigned* bp = (const unsigned*)&sWT[(nt * 16 + r) * K + (kc + 1) * 32 + 16 * hi];
#pragma unroll
        for (int j = 0; j < 8; ++j) ub[(kc + 1) & 1].x[j] = bp[j];
      }
      c = __builtin_amdgcn_wmma_f32_16x16x32_bf16(false, afrag[kc], false, as_v16bf(ub[kc & 1]),
                                                  (short)0, c, false, false);
    }
    int col = nt * 16 + r;
#pragma unroll
    for (int v = 0; v < 8; ++v) {
      int rr = row16 + v + 8 * hi;
      if (rr < M) Y[(long)rr * 64 + col] = c[v];
    }
  }
}

// ---------------- GCN pieces ----------------
__global__ void k_deg(const int* __restrict__ dst, const float* __restrict__ em,
                      float* __restrict__ deg, int E) {
  int e = blockIdx.x * blockDim.x + threadIdx.x;
  if (e >= E) return;
  float w = em ? em[e] : 1.0f;
  if (w != 0.0f) atomicAdd(&deg[dst[e]], w);
}

__global__ void k_gcn_agg(const int* __restrict__ src, const int* __restrict__ dst,
                          const float* __restrict__ em, const float* __restrict__ h,
                          const float* __restrict__ deg, float* __restrict__ agg, int E) {
  long t = (long)blockIdx.x * blockDim.x + threadIdx.x;
  long e = t >> 4;               // 16 threads per edge, float4 each
  int  f4 = (int)(t & 15);
  if (e >= E) return;
  float w = em ? em[e] : 1.0f;
  if (w == 0.0f) return;
  int s = src[e], d = dst[e];
  float norm = rsqrtf(deg[s] + 1.0f) * rsqrtf(deg[d] + 1.0f) * w;
  float4 hv = ((const float4*)(h + (long)s * 64))[f4];
  float* ap = agg + (long)d * 64 + f4 * 4;
  atomicAdd(ap + 0, hv.x * norm);
  atomicAdd(ap + 1, hv.y * norm);
  atomicAdd(ap + 2, hv.z * norm);
  atomicAdd(ap + 3, hv.w * norm);
}

__global__ void k_gcn_fin(const float* __restrict__ h, const float* __restrict__ deg,
                          const float* __restrict__ agg, const float* __restrict__ b,
                          float* __restrict__ xo, long n) {
  long t = (long)blockIdx.x * blockDim.x + threadIdx.x;
  if (t >= n * 64) return;
  long i = t >> 6; int f = (int)(t & 63);
  float dv = deg[i] + 1.0f;
  float v = agg[t] + h[t] / dv + b[f];
  xo[t] = fmaxf(v, 0.0f);
}

// ---------------- EdgeConv: per-edge MLP via WMMA + atomic segment-max ---------
// Tile = 16 edges. GEMM1: A(16x128)=[xi, xj-xi] @ Wa(128x128) -> relu -> LDS
// transpose -> GEMM2: (16x128) @ Wb(128x64) -> atomicMax(fenc) scatter by dst.
// Wa^T (32 KB) is staged into LDS by the Tensor Data Mover when available.
__global__ __launch_bounds__(128)
void k_edgeconv(const float* __restrict__ X,
                const int* __restrict__ src, const int* __restrict__ dst,
                const float* __restrict__ em,
                const unsigned short* __restrict__ WaT,   // 128 x 128 bf16 (row = mid)
                const float* __restrict__ ba,
                const unsigned short* __restrict__ WbT,   // 64 x 128 bf16 (row = out)
                const float* __restrict__ bb,
                unsigned* __restrict__ outEnc, int E) {
  __shared__ unsigned short sWaT[128 * 128];              // 32 KB
  __shared__ unsigned short sMid[4][16 * 128];            // 16 KB (per-wave tiles)
  __shared__ int   sDst[4][16];
  __shared__ float sEm[4][16];

#if HAVE_TDM
  if (threadIdx.x < 32) {
    // Build D# for a 2D 16384x1 bf16 tile: global WaT -> LDS sWaT (TDM, one shot).
    unsigned lds_off = (unsigned)(unsigned long long)(lds_us_t*)&sWaT[0];
    unsigned long long ga = (unsigned long long)WaT;
    tdm_v4u g0;
    g0[0] = 1u;                                   // count=1, user mode
    g0[1] = lds_off;                              // lds_addr (bytes)
    g0[2] = (unsigned)(ga & 0xffffffffu);         // global_addr[31:0]
    g0[3] = (unsigned)((ga >> 32) & 0x1ffffffu) | (2u << 30);  // addr[56:32] | type=2
    tdm_v8i g1 = {};
    g1[0] = (int)(1u << 16);                      // workgroup_mask=0, data_size=1 (2 bytes)
    g1[1] = (int)((16384u & 0xffffu) << 16);      // tensor_dim0[15:0] in bits[31:16]
    g1[2] = (int)((16384u >> 16) | (1u << 16));   // tensor_dim0[31:16] | tensor_dim1[15:0]=1
    g1[3] = (int)(16384u << 16);                  // tensor_dim1[31:16]=0 | tile_dim0=16384
    g1[4] = (int)1;                               // tile_dim1=1, tile_dim2=0 (unused)
    g1[5] = (int)16384;                           // tensor_dim0_stride[31:0]
    g1[6] = 0; g1[7] = 0;                         // stride hi bits / dim1_stride = 0
    tdm_v4i g2 = {}; tdm_v4i g3 = {};
#if defined(__clang_major__) && __clang_major__ >= 23
    tdm_v8i g4 = {};
    __builtin_amdgcn_tensor_load_to_lds(g0, g1, g2, g3, g4, 0);
#else
    __builtin_amdgcn_tensor_load_to_lds(g0, g1, g2, g3, 0);
#endif
    __builtin_amdgcn_s_wait_tensorcnt(0);
  }
#else
  for (int i = threadIdx.x; i < 128 * 128; i += blockDim.x) sWaT[i] = WaT[i];
#endif
  __syncthreads();

  int wave = threadIdx.x >> 5, lane = threadIdx.x & 31;
  int r = lane & 15, hi = lane >> 4;
  int nwaves = (gridDim.x * blockDim.x) >> 5;
  int wid = (blockIdx.x * blockDim.x + threadIdx.x) >> 5;
  int ntiles = (E + 15) >> 4;

  for (int tile = wid; tile < ntiles; tile += nwaves) {
    // prefetch next grid-stride tile's edge indices (global_prefetch_b8)
    int tpf = tile + nwaves;
    if (tpf < ntiles) {
      __builtin_prefetch(&src[tpf * 16 + r], 0, 1);
      __builtin_prefetch(&dst[tpf * 16 + r], 0, 1);
    }
    int e = tile * 16 + r;
    bool ev = e < E;
    int ec = ev ? e : (E - 1);
    int si = src[ec], di = dst[ec];
    float w = em ? em[ec] : 1.0f;
    if (lane < 16) { sDst[wave][r] = di; sEm[wave][r] = ev ? w : 0.0f; }
    const float* xi = X + (long)di * 64;
    const float* xj = X + (long)si * 64;

    // A fragments: feature f<64 -> xi[f]; f>=64 -> xj[f-64]-xi[f-64]
    v16bf a1[4];
#pragma unroll
    for (int kc = 0; kc < 4; ++kc) {
      U8 u;
#pragma unroll
      for (int half = 0; half < 2; ++half) {
        int base = kc * 32 + 16 * hi + half * 8;
#pragma unroll
        for (int j = 0; j < 4; ++j) {
          int f0 = base + 2 * j, f1 = f0 + 1;
          float v0 = (f0 < 64) ? xi[f0] : (xj[f0 - 64] - xi[f0 - 64]);
          float v1 = (f1 < 64) ? xi[f1] : (xj[f1 - 64] - xi[f1 - 64]);
          u.x[half * 4 + j] = pack2(v0, v1);
        }
      }
      a1[kc] = as_v16bf(u);
    }

    // GEMM1 -> relu -> sMid (16 rows x 128 cols, bf16); B frags double-buffered
#pragma unroll
    for (int nt = 0; nt < 8; ++nt) {
      v8f c = {};
      U8 ub[2];
      {
        const unsigned* bp = (const unsigned*)&sWaT[(nt * 16 + r) * 128 + 16 * hi];
#pragma unroll
        for (int j = 0; j < 8; ++j) ub[0].x[j] = bp[j];
      }
#pragma unroll
      for (int kc = 0; kc < 4; ++kc) {
        if (kc < 3) {
          const unsigned* bp = (const unsigned*)&sWaT[(nt * 16 + r) * 128 + (kc + 1) * 32 + 16 * hi];
#pragma unroll
          for (int j = 0; j < 8; ++j) ub[(kc + 1) & 1].x[j] = bp[j];
        }
        c = __builtin_amdgcn_wmma_f32_16x16x32_bf16(false, a1[kc], false, as_v16bf(ub[kc & 1]),
                                                    (short)0, c, false, false);
      }
      int col = nt * 16 + r;
      float bv = ba[col];
#pragma unroll
      for (int v = 0; v < 8; ++v) {
        int row = v + 8 * hi;
        sMid[wave][row * 128 + col] = f2bf(fmaxf(c[v] + bv, 0.0f));
      }
    }

    // GEMM2: (16 x 128) @ Wb(128 x 64); A' and B double-buffered
#pragma unroll
    for (int nt = 0; nt < 4; ++nt) {
      v8f c = {};
      U8 ua[2], ub[2];
      {
        const unsigned* p0 = (const unsigned*)&sMid[wave][r * 128 + 16 * hi];
        const unsigned* p1 = (const unsigned*)&sMid[wave][r * 128 + 16 * hi + 8];
#pragma unroll
        for (int j = 0; j < 4; ++j) { ua[0].x[j] = p0[j]; ua[0].x[4 + j] = p1[j]; }
        const unsigned* bp = (const unsigned*)&WbT[(nt * 16 + r) * 128 + 16 * hi];
#pragma unroll
        for (int j = 0; j < 8; ++j) ub[0].x[j] = bp[j];
      }
#pragma unroll
      for (int kc = 0; kc < 4; ++kc) {
        if (kc < 3) {
          int kn = kc + 1, bi = kn & 1;
          const unsigned* p0 = (const unsigned*)&sMid[wave][r * 128 + kn * 32 + 16 * hi];
          const unsigned* p1 = (const unsigned*)&sMid[wave][r * 128 + kn * 32 + 16 * hi + 8];
#pragma unroll
          for (int j = 0; j < 4; ++j) { ua[bi].x[j] = p0[j]; ua[bi].x[4 + j] = p1[j]; }
          const unsigned* bp = (const unsigned*)&WbT[(nt * 16 + r) * 128 + kn * 32 + 16 * hi];
#pragma unroll
          for (int j = 0; j < 8; ++j) ub[bi].x[j] = bp[j];
        }
        c = __builtin_amdgcn_wmma_f32_16x16x32_bf16(false, as_v16bf(ua[kc & 1]), false,
                                                    as_v16bf(ub[kc & 1]), (short)0, c, false, false);
      }
      int col = nt * 16 + r;
      float bv = bb[col];
#pragma unroll
      for (int v = 0; v < 8; ++v) {
        int row = v + 8 * hi;
        float wgt = sEm[wave][row];
        if (wgt > 0.0f) {
          int d = sDst[wave][row];
          atomicMax(&outEnc[(long)d * 64 + col], fenc(c[v] + bv));
        }
      }
    }
  }
}

// decode + relu (nodes with no edges stay at fenc(-1e30) -> 0 after relu)
__global__ void k_dec_relu(const unsigned* __restrict__ enc, float* __restrict__ x, long cnt) {
  long t = (long)blockIdx.x * blockDim.x + threadIdx.x;
  if (t < cnt) x[t] = fmaxf(fdec(enc[t]), 0.0f);
}

// ---------------- TopK pooling ----------------
__global__ void k_pnorm(const float* __restrict__ p, float* __restrict__ invn) {
  __shared__ float s[64];
  int t = threadIdx.x;
  s[t] = p[t] * p[t];
  __syncthreads();
  for (int st = 32; st > 0; st >>= 1) { if (t < st) s[t] += s[t + st]; __syncthreads(); }
  if (t == 0) *invn = rsqrtf(s[0]);
}

__global__ void k_score(const float* __restrict__ x, const float* __restrict__ p,
                        const float* __restrict__ invn, float* __restrict__ score,
                        unsigned* __restrict__ key, int n) {
  int i = blockIdx.x * blockDim.x + threadIdx.x;
  if (i >= n) return;
  const float* xr = x + (long)i * 64;
  float acc = 0.0f;
#pragma unroll
  for (int j = 0; j < 64; ++j) acc += xr[j] * p[j];
  float s = acc * (*invn);
  score[i] = s;
  key[i] = fenc(s);
}

__global__ void k_hist_hi(const unsigned* __restrict__ key, unsigned* __restrict__ hist, int n) {
  int i = blockIdx.x * blockDim.x + threadIdx.x;
  if (i < n) atomicAdd(&hist[key[i] >> 16], 1u);
}
__global__ void k_hist_lo(const unsigned* __restrict__ key, const unsigned* __restrict__ selHi,
                          unsigned* __restrict__ hist, int n) {
  int i = blockIdx.x * blockDim.x + threadIdx.x;
  if (i < n && (key[i] >> 16) == *selHi) atomicAdd(&hist[key[i] & 0xffffu], 1u);
}

// descending select over 65536 bins: find bin holding the k-th largest element
__global__ void k_select(const unsigned* __restrict__ hist, int kImm, const int* __restrict__ kPtr,
                         unsigned* __restrict__ selBin, int* __restrict__ remK) {
  __shared__ unsigned part[256];
  int t = threadIdx.x;
  unsigned s = 0;
  for (int j = 0; j < 256; ++j) s += hist[t * 256 + j];
  part[t] = s;
  __syncthreads();
  if (t == 0) {
    int k = kPtr ? *kPtr : kImm;
    int acc = 0, seg = 255;
    for (; seg > 0; --seg) { if (acc + (int)part[seg] >= k) break; acc += (int)part[seg]; }
    int b = 255;
    for (; b > 0; --b) { int c = (int)hist[seg * 256 + b]; if (acc + c >= k) break; acc += c; }
    *selBin = (unsigned)(seg * 256 + b);
    *remK = k - acc;
  }
}

__global__ void k_compact(const float* __restrict__ x, const float* __restrict__ score,
                          const unsigned* __restrict__ key,
                          const unsigned* __restrict__ selHi, const unsigned* __restrict__ selLo,
                          const int* __restrict__ needEqP,
                          int* __restrict__ newid, float* __restrict__ xn,
                          int* __restrict__ ctrs, int n) {
  int i = blockIdx.x * blockDim.x + threadIdx.x;
  if (i >= n) return;
  unsigned T = ((*selHi) << 16) | (*selLo);
  unsigned kk = key[i];
  bool keep = false;
  if (kk > T) keep = true;
  else if (kk == T) keep = (atomicAdd(&ctrs[1], 1) < *needEqP);
  int nid = -1;
  if (keep) {
    nid = atomicAdd(&ctrs[0], 1);
    float tnh = tanhf(score[i]);
    const float4* xp = (const float4*)(x + (long)i * 64);
    float4* op = (float4*)(xn + (long)nid * 64);
#pragma unroll
    for (int j = 0; j < 16; ++j) {
      float4 v = xp[j];
      v.x *= tnh; v.y *= tnh; v.z *= tnh; v.w *= tnh;
      op[j] = v;
    }
  }
  newid[i] = nid;
}

__global__ void k_remap(const int* __restrict__ src, const int* __restrict__ dst,
                        const float* __restrict__ em, const int* __restrict__ newid,
                        int* __restrict__ src2, int* __restrict__ dst2,
                        float* __restrict__ em2, int E) {
  int e = blockIdx.x * blockDim.x + threadIdx.x;
  if (e >= E) return;
  int s = newid[src[e]], d = newid[dst[e]];
  float w = em ? em[e] : 1.0f;
  em2[e] = (s >= 0 && d >= 0) ? w : 0.0f;
  src2[e] = s >= 0 ? s : 0;
  dst2[e] = d >= 0 ? d : 0;
}

// ---------------- readout ----------------
__global__ void k_colsum(const float* __restrict__ x, float* __restrict__ g, int n) {
  int tid = blockIdx.x * blockDim.x + threadIdx.x;   // 64 * 256 threads
  int col = tid & 63, part = tid >> 6;
  float acc = 0.0f;
  for (int row = part; row < n; row += 256) acc += x[(long)row * 64 + col];
  atomicAdd(&g[col], acc);
}

__global__ void k_final(const float* __restrict__ g, const float* __restrict__ fcw,
                        const float* __restrict__ fcb, float* __restrict__ out, float invk) {
  __shared__ float sg[64];
  __shared__ float so[16];
  int t = threadIdx.x;                               // 64 threads
  sg[t] = g[t] * invk;
  __syncthreads();
  if (t < 16) {
    float acc = fcb[t];
    for (int j = 0; j < 64; ++j) acc += sg[j] * fcw[j * 16 + t];
    so[t] = acc;
  }
  __syncthreads();
  if (t == 0) {
    float m = -3.4e38f;
    for (int o = 0; o < 16; ++o) m = fmaxf(m, so[o]);
    float ssum = 0.0f;
    for (int o = 0; o < 16; ++o) ssum += expf(so[o] - m);
    float ls = logf(ssum);
    for (int o = 0; o < 16; ++o) out[o] = so[o] - m - ls;
  }
}

// ---------------- host orchestration ----------------
extern "C" void kernel_launch(void* const* d_in, const int* in_sizes, int n_in,
                              void* d_out, int out_size, void* d_ws, size_t ws_size,
                              hipStream_t stream) {
  (void)in_sizes; (void)n_in; (void)out_size; (void)ws_size;
  const float* x    = (const float*)d_in[0];
  const int*   ei   = (const int*)d_in[1];
  const int*   src  = ei;
  const int*   dst  = ei + E_EDGES;
  const float* W1   = (const float*)d_in[3];
  const float* b1   = (const float*)d_in[4];
  const float* e1wa = (const float*)d_in[5];
  const float* e1ba = (const float*)d_in[6];
  const float* e1wb = (const float*)d_in[7];
  const float* e1bb = (const float*)d_in[8];
  const float* p1   = (const float*)d_in[9];
  const float* W2   = (const float*)d_in[10];
  const float* b2   = (const float*)d_in[11];
  const float* e2wa = (const float*)d_in[12];
  const float* e2ba = (const float*)d_in[13];
  const float* e2wb = (const float*)d_in[14];
  const float* e2bb = (const float*)d_in[15];
  const float* p2   = (const float*)d_in[16];
  const float* fcw  = (const float*)d_in[17];
  const float* fcb  = (const float*)d_in[18];
  float* outp = (float*)d_out;

  // carve workspace
  char* wp = (char*)d_ws;
  auto carve = [&](size_t bytes) -> void* {
    void* p = (void*)wp;
    wp += (bytes + 255) & ~(size_t)255;
    return p;
  };
  float*    h     = (float*)carve((size_t)N_NODES * 64 * 4);
  float*    agg   = (float*)carve((size_t)N_NODES * 64 * 4);
  float*    xA    = (float*)carve((size_t)N_NODES * 64 * 4);
  float*    xB    = (float*)carve((size_t)N_NODES * 64 * 4);
  unsigned* enc   = (unsigned*)carve((size_t)N_NODES * 64 * 4);
  float*    deg   = (float*)carve((size_t)N_NODES * 4);
  float*    score = (float*)carve((size_t)N_NODES * 4);
  unsigned* key   = (unsigned*)carve((size_t)N_NODES * 4);
  int*      newid = (int*)carve((size_t)N_NODES * 4);
  float*    xn1   = (float*)carve((size_t)K1_KEEP * 64 * 4);
  float*    xn2   = (float*)carve((size_t)K2_KEEP * 64 * 4);
  int*      src2  = (int*)carve((size_t)E_EDGES * 4);
  int*      dst2  = (int*)carve((size_t)E_EDGES * 4);
  float*    em2   = (float*)carve((size_t)E_EDGES * 4);
  unsigned short* W1T  = (unsigned short*)carve(64 * 128 * 2);
  unsigned short* W2T  = (unsigned short*)carve(64 * 64 * 2);
  unsigned short* Wa1T = (unsigned short*)carve(128 * 128 * 2);
  unsigned short* Wb1T = (unsigned short*)carve(64 * 128 * 2);
  unsigned short* Wa2T = (unsigned short*)carve(128 * 128 * 2);
  unsigned short* Wb2T = (unsigned short*)carve(64 * 128 * 2);
  unsigned* hist   = (unsigned*)carve(65536 * 4);
  unsigned* histLo = (unsigned*)carve(65536 * 4);
  float*    g      = (float*)carve(64 * 4);
  float*    invn   = (float*)carve(4);
  unsigned* selHi  = (unsigned*)carve(4);
  int*      remHi  = (int*)carve(4);
  unsigned* selLo  = (unsigned*)carve(4);
  int*      remLo  = (int*)carve(4);
  int*      ctrs   = (int*)carve(8);

  unsigned negenc;
  { float f = -1e30f; unsigned u; memcpy(&u, &f, 4); negenc = (u & 0x80000000u) ? ~u : (u | 0x80000000u); }

  auto cdiv = [](long a, long b) -> unsigned { return (unsigned)((a + b - 1) / b); };

  // ---- weight prep (fp32 -> transposed bf16) ----
  k_transpose_bf16<<<cdiv(128 * 64, 256), 256, 0, stream>>>(W1, W1T, 128, 64);
  k_transpose_bf16<<<cdiv(64 * 64, 256), 256, 0, stream>>>(W2, W2T, 64, 64);
  k_transpose_bf16<<<cdiv(128 * 128, 256), 256, 0, stream>>>(e1wa, Wa1T, 128, 128);
  k_transpose_bf16<<<cdiv(128 * 64, 256), 256, 0, stream>>>(e1wb, Wb1T, 128, 64);
  k_transpose_bf16<<<cdiv(128 * 128, 256), 256, 0, stream>>>(e2wa, Wa2T, 128, 128);
  k_transpose_bf16<<<cdiv(128 * 64, 256), 256, 0, stream>>>(e2wb, Wb2T, 128, 64);

  // ---- GCN layer 1 ----
  k_fillu<<<cdiv(N_NODES, 256), 256, 0, stream>>>((unsigned*)deg, 0u, N_NODES);
  k_deg<<<cdiv(E_EDGES, 256), 256, 0, stream>>>(dst, nullptr, deg, E_EDGES);
  k_gemm_node<128><<<cdiv(N_NODES, 64), 128, 0, stream>>>(x, W1T, h, N_NODES);
  k_fillu<<<cdiv((long)N_NODES * 64, 256), 256, 0, stream>>>((unsigned*)agg, 0u, (long)N_NODES * 64);
  k_gcn_agg<<<cdiv((long)E_EDGES * 16, 256), 256, 0, stream>>>(src, dst, nullptr, h, deg, agg, E_EDGES);
  k_gcn_fin<<<cdiv((long)N_NODES * 64, 256), 256, 0, stream>>>(h, deg, agg, b1, xA, N_NODES);

  // ---- EdgeConv layer 1 ----
  k_fillu<<<cdiv((long)N_NODES * 64, 256), 256, 0, stream>>>(enc, negenc, (long)N_NODES * 64);
  k_edgeconv<<<1024, 128, 0, stream>>>(xA, src, dst, nullptr, Wa1T, e1ba, Wb1T, e1bb, enc, E_EDGES);
  k_dec_relu<<<cdiv((long)N_NODES * 64, 256), 256, 0, stream>>>(enc, xB, (long)N_NODES * 64);

  // ---- TopK pool 1 ----
  k_pnorm<<<1, 64, 0, stream>>>(p1, invn);
  k_score<<<cdiv(N_NODES, 256), 256, 0, stream>>>(xB, p1, invn, score, key, N_NODES);
  k_fillu<<<cdiv(65536, 256), 256, 0, stream>>>(hist, 0u, 65536);
  k_hist_hi<<<cdiv(N_NODES, 256), 256, 0, stream>>>(key, hist, N_NODES);
  k_select<<<1, 256, 0, stream>>>(hist, K1_KEEP, nullptr, selHi, remHi);
  k_fillu<<<cdiv(65536, 256), 256, 0, stream>>>(histLo, 0u, 65536);
  k_hist_lo<<<cdiv(N_NODES, 256), 256, 0, stream>>>(key, selHi, histLo, N_NODES);
  k_select<<<1, 256, 0, stream>>>(histLo, 0, remHi, selLo, remLo);
  k_fillu<<<1, 256, 0, stream>>>((unsigned*)ctrs, 0u, 2);
  k_compact<<<cdiv(N_NODES, 256), 256, 0, stream>>>(xB, score, key, selHi, selLo, remLo,
                                                    newid, xn1, ctrs, N_NODES);
  k_remap<<<cdiv(E_EDGES, 256), 256, 0, stream>>>(src, dst, nullptr, newid, src2, dst2, em2, E_EDGES);

  // ---- GCN layer 2 ----
  k_fillu<<<cdiv(K1_KEEP, 256), 256, 0, stream>>>((unsigned*)deg, 0u, K1_KEEP);
  k_deg<<<cdiv(E_EDGES, 256), 256, 0, stream>>>(dst2, em2, deg, E_EDGES);
  k_gemm_node<64><<<cdiv(K1_KEEP, 64), 128, 0, stream>>>(xn1, W2T, h, K1_KEEP);
  k_fillu<<<cdiv((long)K1_KEEP * 64, 256), 256, 0, stream>>>((unsigned*)agg, 0u, (long)K1_KEEP * 64);
  k_gcn_agg<<<cdiv((long)E_EDGES * 16, 256), 256, 0, stream>>>(src2, dst2, em2, h, deg, agg, E_EDGES);
  k_gcn_fin<<<cdiv((long)K1_KEEP * 64, 256), 256, 0, stream>>>(h, deg, agg, b2, xA, K1_KEEP);

  // ---- EdgeConv layer 2 ----
  k_fillu<<<cdiv((long)K1_KEEP * 64, 256), 256, 0, stream>>>(enc, negenc, (long)K1_KEEP * 64);
  k_edgeconv<<<1024, 128, 0, stream>>>(xA, src2, dst2, em2, Wa2T, e2ba, Wb2T, e2bb, enc, E_EDGES);
  k_dec_relu<<<cdiv((long)K1_KEEP * 64, 256), 256, 0, stream>>>(enc, xB, (long)K1_KEEP * 64);

  // ---- TopK pool 2 (edges no longer needed afterwards) ----
  k_pnorm<<<1, 64, 0, stream>>>(p2, invn);
  k_score<<<cdiv(K1_KEEP, 256), 256, 0, stream>>>(xB, p2, invn, score, key, K1_KEEP);
  k_fillu<<<cdiv(65536, 256), 256, 0, stream>>>(hist, 0u, 65536);
  k_hist_hi<<<cdiv(K1_KEEP, 256), 256, 0, stream>>>(key, hist, K1_KEEP);
  k_select<<<1, 256, 0, stream>>>(hist, K2_KEEP, nullptr, selHi, remHi);
  k_fillu<<<cdiv(65536, 256), 256, 0, stream>>>(histLo, 0u, 65536);
  k_hist_lo<<<cdiv(K1_KEEP, 256), 256, 0, stream>>>(key, selHi, histLo, K1_KEEP);
  k_select<<<1, 256, 0, stream>>>(histLo, 0, remHi, selLo, remLo);
  k_fillu<<<1, 256, 0, stream>>>((unsigned*)ctrs, 0u, 2);
  k_compact<<<cdiv(K1_KEEP, 256), 256, 0, stream>>>(xB, score, key, selHi, selLo, remLo,
                                                    newid, xn2, ctrs, K1_KEEP);

  // ---- global mean pool + fc + log_softmax ----
  k_fillu<<<1, 64, 0, stream>>>((unsigned*)g, 0u, 64);
  k_colsum<<<64, 256, 0, stream>>>(xn2, g, K2_KEEP);
  k_final<<<1, 64, 0, stream>>>(g, fcw, fcb, outp, 1.0f / (float)K2_KEEP);
}
